// FeatureInteraction_12283606468241
// MI455X (gfx1250) — matrix operand order repaired and checked by reading, not compiled, not run on hardware
//
#include <hip/hip_runtime.h>

// DLRM feature interaction for MI455X (gfx1250, wave32, WMMA).
// out[b] = concat( sparse[b] (3328), triu(emb·embT, k=1) (325), dense[b] (128) )

#define NF      26
#define EMB     128
#define SPW     (NF * EMB)            // 3328
#define NPAIR   (NF * (NF - 1) / 2)   // 325
#define DNW     128
#define OUTW    (SPW + NPAIR + DNW)   // 3781
#define WAVES   4
#define PADROWS 32
#define RS      (EMB + 8)             // LDS row stride in halves (272B, breaks bank conflicts)

typedef __attribute__((ext_vector_type(16))) _Float16 v16h;
typedef __attribute__((ext_vector_type(8)))  _Float16 v8h;
typedef __attribute__((ext_vector_type(4)))  _Float16 v4h;
typedef __attribute__((ext_vector_type(8)))  float    v8f;
typedef __attribute__((ext_vector_type(4)))  float    v4f;

union V16H { v16h v; v8h h[2]; };

// A fragment (16x32 f16, MxK): lane m=lane&15; lanes 0-15 hold K {kb+0..7, kb+16..23},
// lanes 16-31 hold K {kb+8..15, kb+24..31}  (ISA 7.12.2, 16-bit A layout)
__device__ inline v16h load_a_frag(const _Float16* sm, int rowBase, int kb, int lane) {
    const int m = lane & 15;
    const _Float16* p = sm + (rowBase + m) * RS + kb + ((lane < 16) ? 0 : 8);
    V16H r;
    r.h[0] = *(const v8h*)(p);        // first 8 K values
    r.h[1] = *(const v8h*)(p + 16);   // second 8 K values (K+16)
    return r.v;
}

// B fragment (32x16 f16, KxN), B = embT so B[k][n] = emb[rowBase+n][k]:
// lane n=lane&15; lanes 0-15 hold K {kb+0..7, kb+16..23}, lanes 16-31 hold K {kb+8..15, kb+24..31}
__device__ inline v16h load_b_frag(const _Float16* sm, int rowBase, int kb, int lane) {
    const int n = lane & 15;
    const _Float16* p = sm + (rowBase + n) * RS + kb + ((lane < 16) ? 0 : 8);
    V16H r;
    r.h[0] = *(const v8h*)(p);
    r.h[1] = *(const v8h*)(p + 16);
    return r.v;
}

// np.triu_indices(26, k=1) flat index for pair (f, g), f < g
__device__ inline int pair_idx(int f, int g) {
    return f * (2 * NF - 1 - f) / 2 + (g - f - 1);  // f*(51-f)/2 + g - f - 1
}

__global__ __launch_bounds__(WAVES * 32)
void dlrm_interact_kernel(const float* __restrict__ sparse,
                          const float* __restrict__ dense,
                          float* __restrict__ out, int batch) {
    __shared__ __align__(16) _Float16 smem[WAVES][PADROWS * RS];
    const int lane = threadIdx.x & 31;
    const int wave = threadIdx.x >> 5;
    const int row  = blockIdx.x * WAVES + wave;
    const bool active = row < batch;
    _Float16* sm = smem[wave];

    if (active) {
        const float* srow = sparse + (size_t)row * SPW;
        const float* drow = dense  + (size_t)row * DNW;
        float*       orow = out    + (size_t)row * OUTW;

        // zero the padding rows 26..31 (read by the f>=26 WMMA tiles)
        {
            v4h z = {};
            #pragma unroll
            for (int r = NF; r < PADROWS; ++r)
                *(v4h*)(sm + r * RS + lane * 4) = z;
        }

        // Stream the sparse row once: fp32 passthrough copy to out + fp16 convert to LDS.
        // 3328 floats = 832 aligned float4 loads; 26 iterations per lane.
        for (int i = lane; i < SPW / 4; i += 32) {
            const int e = i * 4;              // element offset within row
            v4f d = *(const v4f*)(srow + e);  // 16B-aligned (row base = 13312B multiple)
            // output base is only 4B aligned (3781 odd) -> scalar stores
            orow[e + 0] = d.x;
            orow[e + 1] = d.y;
            orow[e + 2] = d.z;
            orow[e + 3] = d.w;
            const int f = e >> 7;             // feature = e / 128
            const int c = e & 127;            // dim     = e % 128
            v4h h;
            h.x = (_Float16)d.x; h.y = (_Float16)d.y;
            h.z = (_Float16)d.z; h.w = (_Float16)d.w;
            *(v4h*)(sm + f * RS + c) = h;     // 8B-aligned LDS store
        }

        // dense passthrough: 128 floats, 4 per lane
        {
            v4f d = *(const v4f*)(drow + lane * 4);
            float* o = orow + SPW + NPAIR + lane * 4;
            o[0] = d.x; o[1] = d.y; o[2] = d.z; o[3] = d.w;
        }
    }

    __syncthreads();
    if (!active) return;

    // gram = E * E^T over K=128 in 4 WMMA K-steps; symmetric -> tiles T00, T01, T11 only.
    v8f acc00 = {}, acc01 = {}, acc11 = {};
    #pragma unroll
    for (int kb = 0; kb < EMB; kb += 32) {
        v16h a0 = load_a_frag(sm,  0, kb, lane);   // rows 0..15 as A
        v16h a1 = load_a_frag(sm, 16, kb, lane);   // rows 16..31 as A (26+ are zero)
        v16h b0 = load_b_frag(sm,  0, kb, lane);   // rows 0..15 as B (transposed)
        v16h b1 = load_b_frag(sm, 16, kb, lane);   // rows 16..31 as B
        acc00 = __builtin_amdgcn_wmma_f32_16x16x32_f16(false, a0, false, b0,
                                                       (short)0, acc00, false, false);
        acc01 = __builtin_amdgcn_wmma_f32_16x16x32_f16(false, a0, false, b1,
                                                       (short)0, acc01, false, false);
        acc11 = __builtin_amdgcn_wmma_f32_16x16x32_f16(false, a1, false, b1,
                                                       (short)0, acc11, false, false);
    }

    // Scatter the upper triangle. C/D layout: VGPR v, lane l -> M = v + (l<16?0:8), N = l&15.
    float* irow = out + (size_t)row * OUTW + SPW;
    const int mBase = (lane < 16) ? 0 : 8;
    const int n = lane & 15;
    #pragma unroll
    for (int v = 0; v < 8; ++v) {
        {   // T00: f in 0..15, g in 0..15
            const int f = mBase + v, g = n;
            if (f < g) irow[pair_idx(f, g)] = acc00[v];
        }
        {   // T01: f in 0..15, g in 16..25 (f < g always)
            const int f = mBase + v, g = 16 + n;
            if (g < NF) irow[pair_idx(f, g)] = acc01[v];
        }
        {   // T11: f in 16..25, g in 16..25
            const int f = 16 + mBase + v, g = 16 + n;
            if (f < g && g < NF) irow[pair_idx(f, g)] = acc11[v];
        }
    }
}

extern "C" void kernel_launch(void* const* d_in, const int* in_sizes, int n_in,
                              void* d_out, int out_size, void* d_ws, size_t ws_size,
                              hipStream_t stream) {
    const float* sparse = (const float*)d_in[0];
    const float* dense  = (const float*)d_in[1];
    float* out = (float*)d_out;
    const int batch = in_sizes[0] / SPW;
    const int blocks = (batch + WAVES - 1) / WAVES;
    hipLaunchKernelGGL(dlrm_interact_kernel, dim3(blocks), dim3(WAVES * 32), 0, stream,
                       sparse, dense, out, batch);
}
